// LengthRegulator_39659728011758
// MI455X (gfx1250) — matrix operand order repaired and checked by reading, not compiled, not run on hardware
//
#include <hip/hip_runtime.h>

typedef __attribute__((ext_vector_type(16))) _Float16 v16h;
typedef __attribute__((ext_vector_type(8)))  _Float16 v8h;
typedef __attribute__((ext_vector_type(8)))  float    v8f;

#define BB   32
#define TT   512
#define DD   256
#define FF   256
#define MM   3584
#define K3   768            // 3*DD im2col reduction
#define NKB  24             // K3/32 wmma K-steps
#define NTIL 16             // FF/16 N tiles
#define WFRAG_ELEMS (NTIL*NKB*32*16)   // 196608 halves per conv

// ---------------- small prep kernels ----------------

__global__ __launch_bounds__(256) void cvt_f32_f16_kernel(
    const float* __restrict__ in, _Float16* __restrict__ out, int n) {
  int i = blockIdx.x * 256 + threadIdx.x;
  if (i < n) out[i] = (_Float16)in[i];
}

// Pre-swizzle conv weights into this kernel's WMMA B-fragment order (K contiguous per lane half):
// j = ((nt*NKB + kb)*32 + lane)*16 + h  holds  w[n][c][kk] as f16 where
// n = nt*16 + (lane&15), K = kb*32 + ((lane>=16)?16:0) + h, kk = K/256, c = K%256.
__global__ __launch_bounds__(256) void prep_wfrag_kernel(
    const float* __restrict__ w1, const float* __restrict__ w2,
    _Float16* __restrict__ wf1, _Float16* __restrict__ wf2) {
  int i = blockIdx.x * 256 + threadIdx.x;         // over 2*WFRAG_ELEMS
  int conv2 = (i >= WFRAG_ELEMS);
  int j = conv2 ? i - WFRAG_ELEMS : i;
  int h    = j & 15;
  int lane = (j >> 4) & 31;
  int g    = j >> 9;                // nt*NKB + kb
  int kb   = g % NKB;
  int nt   = g / NKB;
  int n  = nt * 16 + (lane & 15);
  int K  = kb * 32 + ((lane >> 4) << 4) + h;
  int kk = K >> 8;
  int c  = K & 255;
  const float* w = conv2 ? w2 : w1;
  float v = w[(n * 256 + c) * 3 + kk];
  (conv2 ? wf2 : wf1)[j] = (_Float16)v;
}

__global__ void cumsum_kernel(const int* __restrict__ target, int* __restrict__ cum) {
  int b = threadIdx.x;
  if (b < BB) {
    int s = 0;
    for (int t = 0; t < TT; ++t) { s += target[b * TT + t]; cum[b * TT + t] = s; }
  }
}

// ---------------- length regulator (bandwidth path) ----------------

__global__ __launch_bounds__(64) void length_regulate_kernel(
    const float* __restrict__ x, const int* __restrict__ cum,
    float* __restrict__ out) {
  int p = blockIdx.x;          // mel frame
  int b = blockIdx.y;          // batch
  const int* cb = cum + b * TT;
  int total = cb[TT - 1];
  // searchsorted(cum, p, side='right')
  int lo = 0, hi = TT;
  while (lo < hi) { int mid = (lo + hi) >> 1; if (cb[mid] <= p) lo = mid + 1; else hi = mid; }
  int idx = lo < TT ? lo : TT - 1;
  float m = (p < total) ? 1.f : 0.f;
  const float4* src = (const float4*)(x + ((size_t)b * TT + idx) * DD);
  float4* dst = (float4*)(out + ((size_t)b * MM + p) * DD);
  float4 v = src[threadIdx.x];
  v.x *= m; v.y *= m; v.z *= m; v.w *= m;
  dst[threadIdx.x] = v;
}

// ---------------- fused conv1d(K=3) + ReLU + LayerNorm (+ optional linear head) ----------------
// One block = 16 consecutive (b,t) rows x full F=256 cols. 8 waves, 2 N-tiles each,
// 24 K-steps of v_wmma_f32_16x16x32_f16 per tile.

__global__ __launch_bounds__(256) void conv_ln_kernel(
    const _Float16* __restrict__ xin,   // [B,T,D] f16 activations
    const _Float16* __restrict__ wf,    // pre-swizzled B fragments
    const float* __restrict__ bias,     // [F]
    const float* __restrict__ gam,      // [F]
    const float* __restrict__ bet,      // [F]
    _Float16* __restrict__ hout,        // [B,T,F] f16 normalized output
    const float* __restrict__ lin_w,    // [F] or null
    const float* __restrict__ lin_b,    // [1] or null
    float* __restrict__ dur_out)        // [B,T] or null
{
  __shared__ _Float16 xs[18 * 256];     // im2col window rows t0-1 .. t0+16
  __shared__ float    hs[16 * 256];     // f32 slab for LN
  __shared__ float    psum[256];
  __shared__ float    psq[256];
  __shared__ float    stats[32];        // per-row mu, rstd

  const int tid  = threadIdx.x;
  const int slab = blockIdx.x;          // 0..1023
  const int b    = slab >> 5;           // 32 slabs per batch
  const int t0   = (slab & 31) << 4;

  // stage activation window into LDS (zero-pad sequence boundaries)
  const _Float16* xb = xin + (size_t)b * TT * DD;
  for (int i = tid; i < 18 * 256; i += 256) {
    int row = i >> 8;
    int c   = i & 255;
    int t   = t0 - 1 + row;
    _Float16 v = (_Float16)0.f;
    if (t >= 0 && t < TT) v = xb[t * DD + c];
    xs[i] = v;
  }
  __syncthreads();

  const int wid   = tid >> 5;
  const int lane  = tid & 31;
  const int mrow  = lane & 15;
  const int abase = (lane >> 4) << 4;   // A-fragment half-K offset: 0 or 16

  v8f acc0 = {}; v8f acc1 = {};
  const int nt0 = wid * 2, nt1 = wid * 2 + 1;
  for (int kb = 0; kb < NKB; ++kb) {
    const int kk = kb >> 3;             // conv tap 0..2
    const int c0 = (kb & 7) << 5;       // channel base within tap
    const _Float16* ap = &xs[(mrow + kk) * 256 + c0 + abase];
    v16h a = *(const v16h*)ap;          // K 0..15 (+abase)
    const v16h bf0 = *(const v16h*)&wf[(size_t)((nt0 * NKB + kb) * 32 + lane) * 16];
    const v16h bf1 = *(const v16h*)&wf[(size_t)((nt1 * NKB + kb) * 32 + lane) * 16];
    acc0 = __builtin_amdgcn_wmma_f32_16x16x32_f16(false, a, false, bf0,
                                                  (short)0, acc0, false, false);
    acc1 = __builtin_amdgcn_wmma_f32_16x16x32_f16(false, a, false, bf1,
                                                  (short)0, acc1, false, false);
  }

  // scatter relu(acc + bias) into hs per C/D layout (lane: col N, VGPR r: row)
  {
    const int roff = (lane >> 4) << 3;
    const int n0 = wid * 32 + (lane & 15);
    const int n1 = n0 + 16;
    const float bi0 = bias[n0], bi1 = bias[n1];
#pragma unroll
    for (int r = 0; r < 8; ++r) {
      float v0 = acc0[r] + bi0; v0 = v0 > 0.f ? v0 : 0.f;
      float v1 = acc1[r] + bi1; v1 = v1 > 0.f ? v1 : 0.f;
      hs[(roff + r) * 256 + n0] = v0;
      hs[(roff + r) * 256 + n1] = v1;
    }
  }
  __syncthreads();

  // layernorm over F=256 per row: two-level LDS reduction
  {
    int row = tid >> 4, seg = tid & 15;
    float s = 0.f, sq = 0.f;
#pragma unroll
    for (int j = 0; j < 16; ++j) {
      float v = hs[row * 256 + seg * 16 + j];
      s += v; sq += v * v;
    }
    psum[tid] = s; psq[tid] = sq;
  }
  __syncthreads();
  if (tid < 16) {
    float s = 0.f, sq = 0.f;
#pragma unroll
    for (int j = 0; j < 16; ++j) { s += psum[tid * 16 + j]; sq += psq[tid * 16 + j]; }
    float mu  = s * (1.f / 256.f);
    float var = sq * (1.f / 256.f) - mu * mu;
    stats[tid * 2]     = mu;
    stats[tid * 2 + 1] = rsqrtf(var + 1e-5f);
  }
  __syncthreads();
  {
    int row = tid >> 4, seg = tid & 15;
    float mu = stats[row * 2], rs = stats[row * 2 + 1];
    size_t obase = (size_t)b * TT * DD + (size_t)(t0 + row) * DD;
#pragma unroll
    for (int j = 0; j < 16; ++j) {
      int c = seg * 16 + j;
      float v = (hs[row * 256 + c] - mu) * rs * gam[c] + bet[c];
      hs[row * 256 + c] = v;
      hout[obase + c] = (_Float16)v;
    }
  }

  if (dur_out) {  // fused linear head + relu (conv2 pass only)
    __syncthreads();
    int row = tid >> 4, seg = tid & 15;
    float s = 0.f;
#pragma unroll
    for (int j = 0; j < 16; ++j) { int c = seg * 16 + j; s += hs[row * 256 + c] * lin_w[c]; }
    psum[tid] = s;
    __syncthreads();
    if (tid < 16) {
      float s2 = 0.f;
#pragma unroll
      for (int j = 0; j < 16; ++j) s2 += psum[tid * 16 + j];
      s2 += lin_b[0];
      dur_out[(size_t)b * TT + t0 + tid] = s2 > 0.f ? s2 : 0.f;
    }
  }
}

// ---------------- host launcher ----------------

extern "C" void kernel_launch(void* const* d_in, const int* in_sizes, int n_in,
                              void* d_out, int out_size, void* d_ws, size_t ws_size,
                              hipStream_t stream) {
  (void)in_sizes; (void)n_in; (void)out_size; (void)ws_size;
  const float* x       = (const float*)d_in[0];
  const float* conv1_w = (const float*)d_in[1];
  const float* conv1_b = (const float*)d_in[2];
  const float* ln1_g   = (const float*)d_in[3];
  const float* ln1_b   = (const float*)d_in[4];
  const float* conv2_w = (const float*)d_in[5];
  const float* conv2_b = (const float*)d_in[6];
  const float* ln2_g   = (const float*)d_in[7];
  const float* ln2_b   = (const float*)d_in[8];
  const float* lin_w   = (const float*)d_in[9];
  const float* lin_b   = (const float*)d_in[10];
  const int*   target  = (const int*)d_in[11];
  // d_in[12] = mel_max_length (compile-time MM)

  char* ws = (char*)d_ws;
  _Float16* x_h  = (_Float16*)ws;                                  // 8 MB
  _Float16* h1_h = (_Float16*)(ws + (size_t)8  * 1024 * 1024);     // 8 MB
  _Float16* w1f  = (_Float16*)(ws + (size_t)16 * 1024 * 1024);     // 384 KB
  _Float16* w2f  = (_Float16*)(ws + (size_t)16 * 1024 * 1024 + 512 * 1024);
  int*      cum  = (int*)     (ws + (size_t)17 * 1024 * 1024);     // 64 KB

  float* out = (float*)d_out;                     // [B, M, D]
  float* dur = out + (size_t)BB * MM * DD;        // [B, T]

  cvt_f32_f16_kernel<<<(BB * TT * DD) / 256, 256, 0, stream>>>(x, x_h, BB * TT * DD);
  prep_wfrag_kernel<<<(2 * WFRAG_ELEMS) / 256, 256, 0, stream>>>(conv1_w, conv2_w, w1f, w2f);
  cumsum_kernel<<<1, 32, 0, stream>>>(target, cum);
  length_regulate_kernel<<<dim3(MM, BB), 64, 0, stream>>>(x, cum, out);
  conv_ln_kernel<<<(BB * TT) / 16, 256, 0, stream>>>(
      x_h, w1f, conv1_b, ln1_g, ln1_b, h1_h, nullptr, nullptr, nullptr);
  conv_ln_kernel<<<(BB * TT) / 16, 256, 0, stream>>>(
      h1_h, w2f, conv2_b, ln2_g, ln2_b, /*reuse*/ x_h, lin_w, lin_b, dur);
}
